// GCNLayer_61538291417593
// MI455X (gfx1250) — matrix operand order repaired and checked by reading, not compiled, not run on hardware
//
#include <hip/hip_runtime.h>
#include <hip/hip_bf16.h>

#define N_NODES   100000
#define N_REL     16
#define N_BASIS   4
#define IN_SIZE   128
#define OUT_SIZE  128
#define N_EDGES   100000
#define ROWS_PER_BLOCK 32

typedef __attribute__((ext_vector_type(16))) __bf16 v16bf;
typedef __attribute__((ext_vector_type(8)))  __bf16 v8bf;
typedef __attribute__((ext_vector_type(8)))  float  v8f;
typedef __attribute__((ext_vector_type(4)))  unsigned int v4u;
typedef __attribute__((ext_vector_type(8)))  int v8i_;
typedef __attribute__((ext_vector_type(4)))  int v4i_;

// ---------------------------------------------------------------------------
// Kernel 1: Wt[r][n][k] = W_r[k][n] = sum_b coeff[r,b] * basis[b][k][n], bf16
// n-major so each WMMA lane's B fragment is one contiguous 32-byte K run.
// ---------------------------------------------------------------------------
__global__ void relw_kernel(const float* __restrict__ coeff,   // [16][4]
                            const float* __restrict__ basis,   // [4][128][128]
                            __bf16* __restrict__ wt) {         // [16][128][128] n-major
    int idx = blockIdx.x * blockDim.x + threadIdx.x;           // 16*128*128 total
    int r = idx >> 14;
    int n = (idx >> 7) & 127;
    int k = idx & 127;
    float s = 0.f;
#pragma unroll
    for (int b = 0; b < N_BASIS; ++b)
        s += coeff[r * N_BASIS + b] * basis[(b * IN_SIZE + k) * OUT_SIZE + n];
    wt[idx] = (__bf16)s;
}

// ---------------------------------------------------------------------------
// Kernel 2: out[n][j] = sum_r bias[r][j]
// ---------------------------------------------------------------------------
__global__ void init_out_kernel(const float* __restrict__ bias,  // [16][128]
                                float* __restrict__ out) {
    int idx = blockIdx.x * blockDim.x + threadIdx.x;             // N_NODES*128
    int j = idx & 127;
    float s = 0.f;
#pragma unroll
    for (int r = 0; r < N_REL; ++r) s += bias[r * OUT_SIZE + j];
    out[idx] = s;
}

// ---------------------------------------------------------------------------
// Kernel 3: zero the per-relation aggregation buffer
// ---------------------------------------------------------------------------
__global__ void zero_kernel(float* __restrict__ p) {
    p[blockIdx.x * blockDim.x + threadIdx.x] = 0.f;
}

// ---------------------------------------------------------------------------
// Kernel 4: scatter  agg[dst] += inp[src] * val   (one wave32 per edge)
// ---------------------------------------------------------------------------
__global__ void scatter_kernel(const float* __restrict__ inp,
                               const float* __restrict__ val,
                               const int*   __restrict__ src,
                               const int*   __restrict__ dst,
                               float* __restrict__ agg, int n_edges) {
    int gid  = blockIdx.x * blockDim.x + threadIdx.x;
    int edge = gid >> 5;
    int lane = gid & 31;
    if (edge >= n_edges) return;
    int s  = src[edge];
    int d  = dst[edge];
    float v = val[edge];
    float4 x = ((const float4*)(inp + (long)s * IN_SIZE))[lane];
    float* ap = agg + (long)d * IN_SIZE + lane * 4;
    atomicAdd(ap + 0, x.x * v);
    atomicAdd(ap + 1, x.y * v);
    atomicAdd(ap + 2, x.z * v);
    atomicAdd(ap + 3, x.w * v);
}

// ---------------------------------------------------------------------------
// Kernel 5: out += agg @ W_r  via V_WMMA_F32_16X16X32_BF16.
// Block: 256 thr = 8 waves. Block tile = 32 rows x 128 cols; wave w owns the
// 16-col strip [16w,16w+16) and iterates 2 M-subtiles. K=128 in 4 chunks of 32.
// LDS: Wt (32 KB, filled by the Tensor Data Mover) + A-tile (32x128 bf16, 8 KB).
// Fragment k order: contiguous k = 16*half + s for both A and B (wave32).
// ---------------------------------------------------------------------------
__global__ __launch_bounds__(256) void gemm_kernel(
        const float*  __restrict__ agg,   // [N_NODES][128] f32
        const __bf16* __restrict__ wt,    // [128][128] bf16, n-major: wt[n*128+k]
        float* __restrict__ out) {        // [N_NODES][128] f32, accumulated
    __shared__ __bf16 Bs[OUT_SIZE * IN_SIZE];         // 32 KB (declared first -> LDS off 0)
    __shared__ __bf16 As[ROWS_PER_BLOCK * IN_SIZE];   //  8 KB

    const int tid  = threadIdx.x;
    const int row0 = blockIdx.x * ROWS_PER_BLOCK;

    // --- Tensor Data Mover: DMA the 128x128 bf16 W tile into LDS. ---
    // D# group0: count=1 | lds_addr | global_addr[56:0] | type=2  (§8.3)
    // D# group1: data_size=2B, tensor_dim0/1=128, tile_dim0/1=128, stride0=128 (§8.4)
    if (tid < 32) {                       // one wave issues; TENSORcnt is per-wave
        unsigned long long ga = (unsigned long long)(uintptr_t)wt;
        unsigned int lds = (unsigned int)(uintptr_t)&Bs[0];
        v4u g0 = { 1u,                                   // count=1, user D#
                   lds,                                  // lds_addr (bytes)
                   (unsigned int)(ga & 0xFFFFFFFFu),
                   (unsigned int)((ga >> 32) & 0x1FFFFFFu) | (2u << 30) }; // type=2
        v8i_ g1 = { (int)(1u << 16),      // data_size=1 -> 2 bytes/elem
                    (int)(128u << 16),    // tensor_dim0[15:0]=128 in bits[31:16]
                    (int)(128u << 16),    // tensor_dim1[15:0]=128 in bits[31:16]
                    (int)(128u << 16),    // tile_dim0=128 in bits[31:16]
                    (int)128,             // tile_dim1=128
                    (int)128,             // tensor_dim0_stride=128
                    0, 0 };
        v4i_ gz4 = { 0, 0, 0, 0 };        // groups 2/3 unused (2D tensor)
        v8i_ gz8 = { 0, 0, 0, 0, 0, 0, 0, 0 };
        __builtin_amdgcn_tensor_load_to_lds(g0, g1, gz4, gz4, gz8, /*cpol=*/0);
        __builtin_amdgcn_s_wait_tensorcnt(0);
    }

    // Stage 32x128 A rows, f32 -> bf16: 1024 float4, 4 per thread
#pragma unroll
    for (int i = 0; i < 4; ++i) {
        int e   = tid + i * 256;          // [0,1024): row = e>>5, 4-col group = e&31
        int row = e >> 5, c4 = e & 31;
        float4 a = ((const float4*)(agg + (long)(row0 + row) * IN_SIZE))[c4];
        __bf16* p = &As[row * IN_SIZE + c4 * 4];
        p[0] = (__bf16)a.x; p[1] = (__bf16)a.y; p[2] = (__bf16)a.z; p[3] = (__bf16)a.w;
    }
    __syncthreads();

    const int wave = tid >> 5;
    const int lane = tid & 31;
    const int nsel = lane & 15;
    const int half = lane >> 4;           // 0 or 1
    const int n0   = wave * 16;

    // Load existing out tiles as the C accumulators (f32), 2 M-subtiles.
    v8f c[2];
#pragma unroll
    for (int m = 0; m < 2; ++m) {
        const float* optr = out + (long)(row0 + m * 16) * OUT_SIZE + n0 + nsel;
#pragma unroll
        for (int v = 0; v < 8; ++v) c[m][v] = optr[(v + half * 8) * OUT_SIZE];
    }

#pragma unroll
    for (int kc = 0; kc < 4; ++kc) {
        const int kbase = kc * 32;

        // B fragment (shared by both M-subtiles): contiguous 32 B per lane.
        const v8bf* bp = (const v8bf*)&Bs[(n0 + nsel) * IN_SIZE + kbase + half * 16];
        v8bf blo = bp[0];
        v8bf bhi = bp[1];
        v16bf b;
#pragma unroll
        for (int i = 0; i < 8; ++i) { b[i] = blo[i]; b[i + 8] = bhi[i]; }

#pragma unroll
        for (int m = 0; m < 2; ++m) {
            // A fragment: two contiguous 16 B runs per lane.
            const v8bf* ap = (const v8bf*)
                &As[(m * 16 + nsel) * IN_SIZE + kbase + half * 16];
            v8bf alo = ap[0];             // K = kbase + 16*half + {0..7}
            v8bf ahi = ap[1];             // K = kbase + 16*half + 8 + {0..7}
            v16bf a;
#pragma unroll
            for (int i = 0; i < 8; ++i) { a[i] = alo[i]; a[i + 8] = ahi[i]; }

            c[m] = __builtin_amdgcn_wmma_f32_16x16x32_bf16(
                       false, a, false, b, (short)0, c[m], false, false);
        }
    }

#pragma unroll
    for (int m = 0; m < 2; ++m) {
        float* optr = out + (long)(row0 + m * 16) * OUT_SIZE + n0 + nsel;
#pragma unroll
        for (int v = 0; v < 8; ++v) optr[(v + half * 8) * OUT_SIZE] = c[m][v];
    }
}

// ---------------------------------------------------------------------------
// Host launch
// ---------------------------------------------------------------------------
extern "C" void kernel_launch(void* const* d_in, const int* in_sizes, int n_in,
                              void* d_out, int out_size, void* d_ws, size_t ws_size,
                              hipStream_t stream) {
    const float* inp      = (const float*)d_in[0]; // [100000][128]
    const float* basis    = (const float*)d_in[1]; // [4][128][128]
    const float* coeff    = (const float*)d_in[2]; // [16][4]
    const float* bias     = (const float*)d_in[3]; // [16][128]
    const float* edge_val = (const float*)d_in[4]; // [16][100000]
    const int*   edge_src = (const int*)  d_in[5]; // [16][100000]
    const int*   edge_dst = (const int*)  d_in[6]; // [16][100000]
    float* out = (float*)d_out;                    // [100000][128]

    // Workspace: bf16 Wt for all relations (512 KB), then agg (51.2 MB)
    __bf16* wt  = (__bf16*)d_ws;
    float*  agg = (float*)((char*)d_ws + (size_t)N_REL * OUT_SIZE * IN_SIZE * sizeof(__bf16));

    relw_kernel<<<(N_REL * OUT_SIZE * IN_SIZE) / 256, 256, 0, stream>>>(coeff, basis, wt);
    init_out_kernel<<<(N_NODES * OUT_SIZE) / 256, 256, 0, stream>>>(bias, out);

    for (int r = 0; r < N_REL; ++r) {
        zero_kernel<<<(N_NODES * IN_SIZE) / 256, 256, 0, stream>>>(agg);
        scatter_kernel<<<(N_EDGES * 32 + 255) / 256, 256, 0, stream>>>(
            inp, edge_val + (size_t)r * N_EDGES,
            edge_src + (size_t)r * N_EDGES, edge_dst + (size_t)r * N_EDGES,
            agg, N_EDGES);
        gemm_kernel<<<N_NODES / ROWS_PER_BLOCK, 256, 0, stream>>>(
            agg, wt + (size_t)r * OUT_SIZE * IN_SIZE, out);
    }
}
